// xSLTM_15779709846234
// MI455X (gfx1250) — compile-verified, matrix operand order checked by reading, not run on hardware
//
#include <hip/hip_runtime.h>
#include <hip/hip_bf16.h>

// ---------------------------------------------------------------------------
// xSLTM on MI455X (gfx1250, wave32, WMMA bf16 16x16x32 + TDM weight streaming)
//
// B=64, S=1024, IN=512, H=1024, NCLS=10.
//   * Weights repacked fp32 -> bf16 WMMA B-fragment layout (~29 MB total,
//     L2-resident across the 1024-step recurrence).
//   * Per (step, layer): grid = H/16 WGs; WG g owns h-columns [16g,16g+16);
//     wave w owns batch rows [16w,16w+16); 4 gate accumulators per wave via
//     v_wmma_f32_16x16x32_bf16 over K = Kx (W phase) then H (R phase).
//   * B fragments streamed into LDS by the Tensor Data Mover: one 2-D D#
//     descriptor per K-tile covers all 4 gates (tile 256 dwords x 4 rows,
//     row stride = per-gate fragment stride). Synced with s_wait_tensorcnt.
//     Falls back to direct global b128 loads if the builtin is unavailable.
//   * A (x_t / h_prev fp32) staged via LDS, lane-gathered per the ISA 16-bit
//     A-matrix layout, converted to bf16 (RNE).
//   * Epilogue: stabilized exponential gating with v_exp_f32 (no libm tanh).
// ---------------------------------------------------------------------------

#define BDIM 64
#define SDIM 1024
#define INDIM 512
#define HDIM 1024
#define NCLS 10

typedef __attribute__((ext_vector_type(16))) __bf16 v16bf;
typedef __attribute__((ext_vector_type(8)))  float  v8f;
typedef __attribute__((ext_vector_type(4)))  unsigned int u32x4;
typedef __attribute__((ext_vector_type(4)))  int          i32x4;
typedef __attribute__((ext_vector_type(8)))  int          i32x8;

#if defined(__gfx1250__) && __has_builtin(__builtin_amdgcn_tensor_load_to_lds) && \
    __has_builtin(__builtin_amdgcn_s_wait_tensorcnt)
#define XSLTM_HAS_TDM 1
#else
#define XSLTM_HAS_TDM 0
#endif

union Frag16 {
    v16bf        v;
    unsigned int d[8];
};

static __device__ __forceinline__ unsigned int pack2_bf16(float a, float b) {
    union { float f; unsigned int u; } x, y;
    x.f = a; y.f = b;
    unsigned int lo = ((x.u + 0x7FFFu + ((x.u >> 16) & 1u)) >> 16) & 0xFFFFu;
    unsigned int hi = ((y.u + 0x7FFFu + ((y.u >> 16) & 1u)) >> 16) & 0xFFFFu;
    return lo | (hi << 16);
}

// K-offset of the first of the 2 packed K-values held in A/B VGPR v
// (16-bit element ISA layout, wave32).
static __device__ __forceinline__ int kbase_of(int v, int laneHi) {
    return ((v < 4) ? (2 * v) : (2 * v + 8)) + (laneHi ? 8 : 0);
}

#if XSLTM_HAS_TDM
// ---------------------------------------------------------------------------
// Issue one TDM load: a 2-D tile (tile_dim0 = ndw dwords, tile_dim1 = 4 rows,
// row stride = gate_stride_dw dwords) from global into LDS at lds_off.
// Descriptor per CDNA5 ISA D# Group0/Group1 bitfields (data_size = 4 bytes).
// ---------------------------------------------------------------------------
static __device__ __forceinline__ void tdm_load_b_frags(const void* gsrc,
                                                        unsigned int lds_off,
                                                        unsigned int gate_stride_dw,
                                                        unsigned int ndw) {
    const unsigned long long ga = (unsigned long long)(uintptr_t)gsrc;
    u32x4 g0;
    g0[0] = 1u;                                          // count=1, user D#
    g0[1] = lds_off;                                     // lds_addr (bytes)
    g0[2] = (unsigned int)(ga & 0xFFFFFFFFull);          // global_addr[31:0]
    g0[3] = (unsigned int)((ga >> 32) & 0x1FFFFFFull)    // global_addr[56:32]
          | (2u << 30);                                  // type = 2 ("image")

    const unsigned int T0  = ndw;        // tensor_dim0 (dword units)
    const unsigned int T1  = 4u;         // tensor_dim1 (4 gate rows)
    const unsigned int TD0 = ndw;        // tile_dim0
    const unsigned int TD1 = 4u;         // tile_dim1
    const unsigned long long S0 = gate_stride_dw;        // tensor_dim0_stride

    i32x8 g1;
    g1[0] = (int)(2u << 16);                                       // data_size=4B
    g1[1] = (int)((T0 & 0xFFFFu) << 16);                           // barrier=0 | T0.lo
    g1[2] = (int)(((T0 >> 16) & 0xFFFFu) | ((T1 & 0xFFFFu) << 16));
    g1[3] = (int)(((T1 >> 16) & 0xFFFFu) | (TD0 << 16));
    g1[4] = (int)(TD1 & 0xFFFFu);                                  // tile_dim2=0
    g1[5] = (int)(unsigned int)(S0 & 0xFFFFFFFFull);
    g1[6] = (int)(unsigned int)((S0 >> 32) & 0xFFFFu);             // stride1=0
    g1[7] = 0;

    const i32x4 gz = {0, 0, 0, 0};
#if __clang_major__ >= 23
    const i32x8 gz8 = {0, 0, 0, 0, 0, 0, 0, 0};
    __builtin_amdgcn_tensor_load_to_lds(g0, g1, gz, gz, gz8, 0);
#else
    __builtin_amdgcn_tensor_load_to_lds(g0, g1, gz, gz, 0);
#endif
}
#endif  // XSLTM_HAS_TDM

// ---------------------------------------------------------------------------
// Repack a fp32 row-major [K, N] weight matrix into bf16 WMMA B-fragments.
// Fragment fid = ct * (K/32) + kt  (ct = column tile of 16, kt = K tile of 32).
// Each fragment: 32 lanes x 16 halfs, lane-major (32 contiguous bytes/lane).
// One wave per fragment; block = 128 threads = 4 fragments.
// ---------------------------------------------------------------------------
__global__ __launch_bounds__(128)
void xsltm_repack_bf16(const float* __restrict__ W,
                       unsigned short* __restrict__ out,
                       int K, int N) {
    const int frag   = blockIdx.x * 4 + (threadIdx.x >> 5);
    const int lane   = threadIdx.x & 31;
    const int laneHi = lane >> 4;
    const int ktiles = K >> 5;
    const int ct     = frag / ktiles;
    const int kt     = frag % ktiles;
    const int ncol   = ct * 16 + (lane & 15);

    unsigned int* dst = (unsigned int*)(out + (size_t)frag * 512 + (size_t)lane * 16);
#pragma unroll
    for (int v = 0; v < 8; ++v) {
        const int k = kt * 32 + kbase_of(v, laneHi);
        const float a = W[(size_t)k * N + ncol];
        const float b = W[(size_t)(k + 1) * N + ncol];
        dst[v] = pack2_bf16(a, b);
    }
}

// ---------------------------------------------------------------------------
// One sLSTM layer, one timestep.
//   pre = [xin | hprev] @ [Wp ; Rp] + bias,  then exp-gate update.
// grid.x = HDIM/16 = 64 (column tiles), block = 128 (4 waves of 32).
// ---------------------------------------------------------------------------
__global__ __launch_bounds__(128)
void xsltm_step(const float* __restrict__ xin, int Kx, long long ldx,
                const float* __restrict__ hprev,
                const unsigned short* __restrict__ Wp,
                const unsigned short* __restrict__ Rp,
                const float* __restrict__ bias,
                float* __restrict__ cst, float* __restrict__ nst,
                float* __restrict__ mst, float* __restrict__ hout) {
    __shared__ float ldsA[64 * 36];            // 64 rows x 32 K fp32, padded
#if XSLTM_HAS_TDM
    __shared__ unsigned short ldsB[4 * 512];   // 4 gate fragments x 1 KB
#endif

    const int tid    = threadIdx.x;
    const int wave   = tid >> 5;
    const int lane   = tid & 31;
    const int laneHi = lane >> 4;
    const int lm     = lane & 15;
    const int g      = blockIdx.x;             // h-column tile

    v8f acc[4] = {};                           // z, i, f, o accumulators (f32)

    const int Mrow = wave * 16 + lm;           // A-matrix row this lane gathers

#pragma unroll 1
    for (int phase = 0; phase < 2; ++phase) {
        const float* A           = phase ? hprev : xin;
        const long long lda      = phase ? (long long)HDIM : ldx;
        const unsigned short* Wb = phase ? Rp : Wp;
        const int ktiles         = (phase ? HDIM : Kx) >> 5;
        const size_t ctStride    = (size_t)ktiles * 512;                 // halfs
        const unsigned int gateStrideDw = (unsigned int)((HDIM / 16) * ktiles * 256);

#pragma unroll 1
        for (int kt = 0; kt < ktiles; ++kt) {
            // ---- TDM: stream this K-tile's 4 gate B-fragments into LDS ----
#if XSLTM_HAS_TDM
            if (wave == 0) {
                const unsigned short* bsrc =
                    Wb + (size_t)g * ctStride + (size_t)kt * 512;        // gate 0
                tdm_load_b_frags(bsrc, (unsigned int)(uintptr_t)ldsB,
                                 gateStrideDw, 256u);
            }
#endif
            // ---- stage A block [64 x 32] fp32 into LDS (coalesced b128) ----
            {
                const int row   = tid >> 1;
                const int cbase = (tid & 1) * 16;
                const float* src = A + (size_t)row * lda + (size_t)kt * 32 + cbase;
                float4 p0 = *(const float4*)(src + 0);
                float4 p1 = *(const float4*)(src + 4);
                float4 p2 = *(const float4*)(src + 8);
                float4 p3 = *(const float4*)(src + 12);
                float* d = &ldsA[row * 36 + cbase];
                *(float4*)(d + 0)  = p0;
                *(float4*)(d + 4)  = p1;
                *(float4*)(d + 8)  = p2;
                *(float4*)(d + 12) = p3;
            }
#if XSLTM_HAS_TDM
            if (wave == 0) __builtin_amdgcn_s_wait_tensorcnt(0);
#endif
            __syncthreads();

            // ---- build bf16 A-fragment per ISA 16-bit A layout ----
            Frag16 af;
#pragma unroll
            for (int v = 0; v < 8; ++v) {
                const int kb = kbase_of(v, laneHi);
                float2 p = *(const float2*)&ldsA[Mrow * 36 + kb];
                af.d[v] = pack2_bf16(p.x, p.y);
            }

            // ---- 4 gate tiles: B-fragment from LDS (TDM) or global, WMMA ----
            const size_t fragBase = (size_t)kt * 512 + (size_t)lane * 16;
#pragma unroll
            for (int gate = 0; gate < 4; ++gate) {
                Frag16 bf;
#if XSLTM_HAS_TDM
                const unsigned int* bp32 =
                    (const unsigned int*)(ldsB + gate * 512) + lane * 8;
#pragma unroll
                for (int v = 0; v < 8; ++v) bf.d[v] = bp32[v];
                // keep next K-tile's gate strip warming in L2
                __builtin_prefetch(
                    (const void*)(Wb + (size_t)(gate * (HDIM / 16) + g) * ctStride +
                                  fragBase + 512), 0, 1);
#else
                const unsigned short* bp =
                    Wb + (size_t)(gate * (HDIM / 16) + g) * ctStride + fragBase;
                const unsigned int* bp32 = (const unsigned int*)bp;
#pragma unroll
                for (int v = 0; v < 8; ++v) bf.d[v] = bp32[v];
                __builtin_prefetch((const void*)(bp + 512), 0, 1);
#endif
                acc[gate] = __builtin_amdgcn_wmma_f32_16x16x32_bf16(
                    false, af.v, false, bf.v, (short)0, acc[gate], false, false);
            }
            __syncthreads();   // protect ldsA/ldsB for next K-tile
        }
    }

    // ---- epilogue: stabilized exponential gating (v_exp_f32 only) ----
    const int j = g * 16 + lm;                 // h column
    const float bz  = bias[j];
    const float bi  = bias[HDIM + j];
    const float bfv = bias[2 * HDIM + j];
    const float bo  = bias[3 * HDIM + j];

#pragma unroll
    for (int r = 0; r < 8; ++r) {
        const int brow = wave * 16 + r + laneHi * 8;     // batch row (C layout)
        const size_t idx = (size_t)brow * HDIM + j;
        const float z_p = acc[0][r] + bz;
        const float i_p = acc[1][r] + bi;
        const float f_p = acc[2][r] + bfv;
        const float o_p = acc[3][r] + bo;

        const float cm = cst[idx];
        const float nn = nst[idx];
        const float mm = mst[idx];

        const float e2    = __expf(2.0f * z_p);
        const float z     = 1.0f - 2.0f / (e2 + 1.0f);   // tanh, saturates to +/-1
        const float o     = 1.0f / (1.0f + __expf(-o_p));
        const float m_new = fmaxf(f_p + mm, i_p);
        const float i_g   = __expf(i_p - m_new);
        const float f_g   = __expf(f_p + mm - m_new);
        const float c_new = f_g * cm + i_g * z;
        const float n_new = f_g * nn + i_g;
        const float h_new = o * c_new / n_new;

        cst[idx]  = c_new;
        nst[idx]  = n_new;
        mst[idx]  = m_new;
        hout[idx] = h_new;
    }
}

// ---------------------------------------------------------------------------
// Final classifier: out[b, cls] = h1[b,:] @ Wc[:,cls] + bc[cls]   (64 x 10)
// ---------------------------------------------------------------------------
__global__ __launch_bounds__(64)
void xsltm_classifier(const float* __restrict__ h,
                      const float* __restrict__ Wc,
                      const float* __restrict__ bc,
                      float* __restrict__ out) {
    const int b   = threadIdx.x;   // 0..63
    const int cls = blockIdx.x;    // 0..9
    float s = bc[cls];
    const float* hr = h + (size_t)b * HDIM;
#pragma unroll 4
    for (int k = 0; k < HDIM; ++k) s = fmaf(hr[k], Wc[(size_t)k * NCLS + cls], s);
    out[b * NCLS + cls] = s;
}

// ---------------------------------------------------------------------------
extern "C" void kernel_launch(void* const* d_in, const int* in_sizes, int n_in,
                              void* d_out, int out_size, void* d_ws, size_t ws_size,
                              hipStream_t stream) {
    const float* x  = (const float*)d_in[0];
    const float* W0 = (const float*)d_in[1];
    const float* R0 = (const float*)d_in[2];
    const float* b0 = (const float*)d_in[3];
    const float* W1 = (const float*)d_in[4];
    const float* R1 = (const float*)d_in[5];
    const float* b1 = (const float*)d_in[6];
    const float* Wc = (const float*)d_in[7];
    const float* bc = (const float*)d_in[8];

    char* ws = (char*)d_ws;
    size_t off = 0;
    auto carve = [&](size_t bytes) -> void* {
        void* p = ws + off;
        off += (bytes + 255) & ~(size_t)255;
        return p;
    };

    unsigned short* Wp0 = (unsigned short*)carve((size_t)INDIM * 4 * HDIM * 2);
    unsigned short* Rp0 = (unsigned short*)carve((size_t)HDIM * 4 * HDIM * 2);
    unsigned short* Wp1 = (unsigned short*)carve((size_t)HDIM * 4 * HDIM * 2);
    unsigned short* Rp1 = (unsigned short*)carve((size_t)HDIM * 4 * HDIM * 2);

    const size_t stateElems = (size_t)BDIM * HDIM;       // 64K floats
    float* states = (float*)carve(stateElems * 10 * sizeof(float));
    float* h0[2] = { states + 0 * stateElems, states + 1 * stateElems };
    float* h1[2] = { states + 2 * stateElems, states + 3 * stateElems };
    float* c0 = states + 4 * stateElems;
    float* n0 = states + 5 * stateElems;
    float* m0 = states + 6 * stateElems;
    float* c1 = states + 7 * stateElems;
    float* n1 = states + 8 * stateElems;
    float* m1 = states + 9 * stateElems;

    // zero-init all recurrent state (h, c, n, m for both layers)
    hipMemsetAsync(states, 0, stateElems * 10 * sizeof(float), stream);

    // per-call weight repack fp32 -> bf16 WMMA fragments
    {
        const int fragsW0 = (INDIM / 32) * (4 * HDIM / 16);   // 4096
        const int fragsR  = (HDIM / 32) * (4 * HDIM / 16);    // 8192
        xsltm_repack_bf16<<<fragsW0 / 4, 128, 0, stream>>>(W0, Wp0, INDIM, 4 * HDIM);
        xsltm_repack_bf16<<<fragsR  / 4, 128, 0, stream>>>(R0, Rp0, HDIM,  4 * HDIM);
        xsltm_repack_bf16<<<fragsR  / 4, 128, 0, stream>>>(W1, Wp1, HDIM,  4 * HDIM);
        xsltm_repack_bf16<<<fragsR  / 4, 128, 0, stream>>>(R1, Rp1, HDIM,  4 * HDIM);
    }

    // recurrence: 1024 steps x 2 layers, h double-buffered
    const long long ldx0 = (long long)SDIM * INDIM;   // x row stride
    int cur = 0;
    for (int t = 0; t < SDIM; ++t) {
        const int nxt = cur ^ 1;
        xsltm_step<<<HDIM / 16, 128, 0, stream>>>(
            x + (size_t)t * INDIM, INDIM, ldx0, h0[cur],
            Wp0, Rp0, b0, c0, n0, m0, h0[nxt]);
        xsltm_step<<<HDIM / 16, 128, 0, stream>>>(
            h0[nxt], HDIM, (long long)HDIM, h1[cur],
            Wp1, Rp1, b1, c1, n1, m1, h1[nxt]);
        cur = nxt;
    }

    xsltm_classifier<<<NCLS, BDIM, 0, stream>>>(h1[cur], Wc, bc, (float*)d_out);
}
